// EdgeMaskNet_34342558499148
// MI455X (gfx1250) — matrix operand order repaired and checked by reading, not compiled, not run on hardware
//
#include <hip/hip_runtime.h>

#define NN   100000
#define NE   1600000
#define NB   512
#define NEP  200000
#define KNIN 64
#define HID  72
#define NLAY 3
#define BEPS 1e-5f

typedef __attribute__((ext_vector_type(16))) _Float16 v16h;
typedef __attribute__((ext_vector_type(8)))  float    v8f;

// ---------- helpers ----------
__device__ __forceinline__ unsigned enc_f32(float f) {
    unsigned u = __float_as_uint(f);
    return (u & 0x80000000u) ? ~u : (u | 0x80000000u);
}
__device__ __forceinline__ float dec_f32(unsigned u) {
    unsigned b = (u & 0x80000000u) ? (u & 0x7FFFFFFFu) : ~u;
    return __uint_as_float(b);
}

// ---------- degree / norm ----------
__global__ void k_deg(const int* __restrict__ col, float* __restrict__ deg) {
    int e = blockIdx.x * blockDim.x + threadIdx.x;
    if (e < NE) atomicAdd(&deg[col[e]], 1.0f);
}
__global__ void k_dis(float* __restrict__ deg) {
    int i = blockIdx.x * blockDim.x + threadIdx.x;
    if (i < NN) { float d = deg[i]; deg[i] = (d > 0.f) ? rsqrtf(d) : 0.f; }
}
__global__ void k_norm(const int* __restrict__ row, const int* __restrict__ col,
                       const float* __restrict__ dis, float* __restrict__ nrm) {
    int e = blockIdx.x * blockDim.x + threadIdx.x;
    if (e < NE) nrm[e] = dis[row[e]] * dis[col[e]];
}

// ---------- pack W[K][Ncols] f32 -> WMMA B-fragment order f16 ----------
// Bp layout: [tile(5)][ks(Ksteps)][lane(32)][elem(16)], zero-padded OOB.
// elem e=2q+j <-> kb = ks*32 + (q<4 ? 2q + 8*(lane/16) + j : 16 + 2(q-4) + 8*(lane/16) + j), n = tile*16 + lane%16
__global__ void k_packB(const float* __restrict__ W, _Float16* __restrict__ Bp,
                        int K, int Ncols, int Ksteps, int total) {
    int idx = blockIdx.x * blockDim.x + threadIdx.x;
    if (idx >= total) return;
    int e    = idx & 15;
    int lane = (idx >> 4) & 31;
    int ks   = (idx >> 9) % Ksteps;
    int t    = (idx >> 9) / Ksteps;
    int nl = lane & 15, half = lane >> 4;
    int q = e >> 1, j = e & 1;
    int k = (ks << 5) + ((q < 4) ? (2 * q + 8 * half + j) : (16 + 2 * (q - 4) + 8 * half + j));
    int n = t * 16 + nl;
    float v = (k < K && n < Ncols) ? W[(size_t)k * Ncols + n] : 0.f;
    Bp[idx] = (_Float16)v;
}

// ---------- WMMA GEMM: out = act(A[MxK] @ W[KxHID] (+bias) (+addend)) ----------
// block = 160 threads = 5 waves; 16 rows/block; wave w owns cols [16w,16w+16)
// A staged in LDS fragment-ordered: Af[ks][lane][16] (contiguous 32B per lane)
__global__ __launch_bounds__(160)
void k_gemm16(const float* __restrict__ A, int M, int K, int Ksteps,
              const _Float16* __restrict__ Bp,
              const float* __restrict__ bias,
              const float* __restrict__ addend,
              int do_relu, float* __restrict__ out) {
    __shared__ _Float16 Af[3 * 512];
    const int tid = threadIdx.x;
    const int rowbase = blockIdx.x * 16;
    for (int idx = tid; idx < (Ksteps << 9); idx += 160) {
        int e = idx & 15, lane = (idx >> 4) & 31, ks = idx >> 9;
        int nl = lane & 15, half = lane >> 4;
        int q = e >> 1, j = e & 1;
        int kk = (q < 4) ? (2 * q + 8 * half + j) : (16 + 2 * (q - 4) + 8 * half + j);
        int k = (ks << 5) + kk;
        int row = rowbase + nl;
        float v = (row < M && k < K) ? A[(size_t)row * K + k] : 0.f;
        Af[idx] = (_Float16)v;
    }
    __syncthreads();

    const int w = tid >> 5, lane = tid & 31;
    const int nloc = lane & 15, half = lane >> 4;
    const int col = w * 16 + nloc;
    const v16h* bp = (const v16h*)(Bp + ((size_t)(w * Ksteps) << 9));

    v8f c = {};
#pragma unroll 3
    for (int ks = 0; ks < Ksteps; ++ks) {
        v16h a = *(const v16h*)&Af[(ks << 9) + (lane << 4)];
        v16h b = bp[(ks << 5) + lane];
        c = __builtin_amdgcn_wmma_f32_16x16x32_f16(false, a, false, b, (short)0, c, false, false);
    }

#pragma unroll
    for (int r = 0; r < 8; ++r) {
        int m = r + 8 * half;                    // C layout: lanes16-31 hold M=8..15
        int row = rowbase + m;
        if (row < M && col < HID) {
            float v = c[r];
            if (bias)   v += bias[col];
            if (addend) v += addend[(size_t)row * HID + col];
            if (do_relu) v = fmaxf(v, 0.f);
            out[(size_t)row * HID + col] = v;
        }
    }
}

// ---------- edge scatter: agg[col] += norm * hw[row] (wave per edge) ----------
__global__ __launch_bounds__(256)
void k_edge_scatter(const int* __restrict__ row, const int* __restrict__ col,
                    const float* __restrict__ nrm, const float* __restrict__ hw,
                    float* __restrict__ agg) {
    int e = blockIdx.x * 8 + (threadIdx.x >> 5);
    int lane = threadIdx.x & 31;
    if (e >= NE) return;
    int r = row[e], cdst = col[e];
    float nv = nrm[e];
    const float* src = hw + (size_t)r * HID;
    float* dst = agg + (size_t)cdst * HID;
    for (int j = lane; j < HID; j += 32) atomicAdd(&dst[j], nv * src[j]);
}

// ---------- batchnorm ----------
__global__ __launch_bounds__(256)
void k_bn_stats(const float* __restrict__ h, float* __restrict__ stats) {
    __shared__ float s[HID], ss[HID];
    int tid = threadIdx.x;
    if (tid < HID) { s[tid] = 0.f; ss[tid] = 0.f; }
    __syncthreads();
    int rowbase = blockIdx.x * 64;
    for (int idx = tid; idx < 64 * HID; idx += 256) {
        int r = idx / HID, cdx = idx - r * HID;
        int row = rowbase + r;
        if (row < NN) {
            float v = h[(size_t)row * HID + cdx];
            atomicAdd(&s[cdx], v);
            atomicAdd(&ss[cdx], v * v);
        }
    }
    __syncthreads();
    if (tid < HID) {
        atomicAdd(&stats[tid],       s[tid]);
        atomicAdd(&stats[HID + tid], ss[tid]);
    }
}
__global__ void k_bn_apply(const float* __restrict__ hin, const float* __restrict__ stats,
                           const float* __restrict__ gamma, const float* __restrict__ beta,
                           float* __restrict__ hout) {
    int idx = blockIdx.x * blockDim.x + threadIdx.x;
    if (idx >= NN * HID) return;
    int cdx = idx % HID;
    float mu  = stats[cdx] * (1.0f / NN);
    float var = stats[HID + cdx] * (1.0f / NN) - mu * mu;
    hout[idx] = (hin[idx] - mu) * rsqrtf(var + BEPS) * gamma[cdx] + beta[cdx];
}

// ---------- argmax over assign rows (wave per node) ----------
__global__ __launch_bounds__(256)
void k_argmax(const float* __restrict__ assign, int* __restrict__ bidx) {
    int n = blockIdx.x * 8 + (threadIdx.x >> 5);
    int lane = threadIdx.x & 31;
    if (n >= NN) return;
    float bv = -3.402823466e38f; int bi = 0x7fffffff;
    const float* rowp = assign + (size_t)n * NB;
    for (int j = lane; j < NB; j += 32) {
        float v = rowp[j];
        if (v > bv) { bv = v; bi = j; }
    }
    for (int off = 16; off > 0; off >>= 1) {
        float ov = __shfl_xor(bv, off, 32);
        int   oi = __shfl_xor(bi, off, 32);
        if (ov > bv || (ov == bv && oi < bi)) { bv = ov; bi = oi; }
    }
    if (lane == 0) bidx[n] = bi;
}

// ---------- pooling ----------
__global__ __launch_bounds__(256)
void k_pool(const float* __restrict__ h, const float* __restrict__ ef,
            const int* __restrict__ bidx, float* __restrict__ cnt,
            float* __restrict__ p1, unsigned* __restrict__ p2) {
    int n = blockIdx.x * 8 + (threadIdx.x >> 5);
    int lane = threadIdx.x & 31;
    if (n >= NN) return;
    int b = bidx[n];
    if (lane == 0) atomicAdd(&cnt[b], 1.0f);
    for (int cdx = lane; cdx < 2 * HID; cdx += 32) {
        float v = (cdx < HID) ? h[(size_t)n * HID + cdx] : ef[(size_t)n * HID + (cdx - HID)];
        atomicAdd(&p1[(size_t)b * 2 * HID + cdx], v);
        atomicMax(&p2[(size_t)b * 2 * HID + cdx], enc_f32(v));
    }
}
__global__ void k_xp(const float* __restrict__ cnt, const float* __restrict__ p1,
                     const unsigned* __restrict__ p2, float* __restrict__ xp) {
    int idx = blockIdx.x * blockDim.x + threadIdx.x;
    if (idx >= NB * 6 * HID) return;
    int b = idx / (6 * HID), cdx = idx % (6 * HID);
    float v;
    if (cdx < 2 * HID)      v = p1[b * 2 * HID + cdx];
    else if (cdx < 4 * HID) v = dec_f32(p2[b * 2 * HID + (cdx - 2 * HID)]);
    else                    v = p1[b * 2 * HID + (cdx - 4 * HID)] / fmaxf(cnt[b], 1.0f);
    xp[idx] = v;
}

// ---------- pair MLP: tanh(concat(xp[a],xp[b]) @ W1 + b1) @ w2 + b2 ----------
// block = 160 threads = 5 waves; 16 pairs/block; K = 864 = 27*32
__global__ __launch_bounds__(160)
void k_pair_mlp(const float* __restrict__ xp, const int* __restrict__ pe,
                const _Float16* __restrict__ W1p, const float* __restrict__ b1,
                const float* __restrict__ w2, const float* __restrict__ b2,
                float* __restrict__ out) {
    const int KSTEPS = 27;                     // 864 / 32
    __shared__ _Float16 Pf[27 * 512];          // fragment-ordered A tile
    __shared__ float Hd[16][80];
    __shared__ int pa[16], pb[16];
    const int tid = threadIdx.x;
    const int base = blockIdx.x * 16;
    if (tid < 16) {
        int p  = base + tid;
        int pp = (p < NEP) ? p : 0;
        pa[tid] = pe[pp];
        pb[tid] = pe[NEP + pp];
    }
    __syncthreads();
    for (int idx = tid; idx < (KSTEPS << 9); idx += 160) {
        int e = idx & 15, lane = (idx >> 4) & 31, ks = idx >> 9;
        int nl = lane & 15, half = lane >> 4;
        int q = e >> 1, j = e & 1;
        int kk = (q < 4) ? (2 * q + 8 * half + j) : (16 + 2 * (q - 4) + 8 * half + j);
        int k = (ks << 5) + kk;                // 0..863
        float v = (k < 6 * HID) ? xp[(size_t)pa[nl] * 6 * HID + k]
                                : xp[(size_t)pb[nl] * 6 * HID + (k - 6 * HID)];
        Pf[idx] = (_Float16)v;
    }
    __syncthreads();

    const int w = tid >> 5, lane = tid & 31;
    const int nloc = lane & 15, half = lane >> 4;
    const int col = w * 16 + nloc;
    const v16h* bp = (const v16h*)(W1p + ((size_t)(w * KSTEPS) << 9));

    v8f c = {};
#pragma unroll 3
    for (int ks = 0; ks < KSTEPS; ++ks) {
        v16h a = *(const v16h*)&Pf[(ks << 9) + (lane << 4)];
        v16h b = bp[(ks << 5) + lane];
        c = __builtin_amdgcn_wmma_f32_16x16x32_f16(false, a, false, b, (short)0, c, false, false);
    }

#pragma unroll
    for (int r = 0; r < 8; ++r) {
        int m = r + 8 * half;
        if (col < HID) Hd[m][col] = tanhf(c[r] + b1[col]);
    }
    __syncthreads();
    if (tid < 16) {
        int p = base + tid;
        if (p < NEP) {
            float s = b2[0];
            for (int cdx = 0; cdx < HID; ++cdx) s += Hd[tid][cdx] * w2[cdx];
            out[p] = s;
        }
    }
}

// ---------- host ----------
extern "C" void kernel_launch(void* const* d_in, const int* in_sizes, int n_in,
                              void* d_out, int out_size, void* d_ws, size_t ws_size,
                              hipStream_t stream) {
    (void)in_sizes; (void)n_in; (void)out_size; (void)ws_size;
    const float* x      = (const float*)d_in[0];
    const float* emb    = (const float*)d_in[1];
    const float* assign = (const float*)d_in[2];
    const int*   ei     = (const int*)d_in[3];
    const int*   pe     = (const int*)d_in[4];
    const float* node_w = (const float*)d_in[5];
    const float* node_b = (const float*)d_in[6];
    const float* emb_w  = (const float*)d_in[7];
    const float* emb_b  = (const float*)d_in[8];
    const float* wi     = (const float*)d_in[9];
    const float* wr     = (const float*)d_in[10];
    const float* cb     = (const float*)d_in[11];
    const float* gam    = (const float*)d_in[12];
    const float* bet    = (const float*)d_in[13];
    const float* w1     = (const float*)d_in[14];
    const float* b1     = (const float*)d_in[15];
    const float* w2     = (const float*)d_in[16];
    const float* b2     = (const float*)d_in[17];
    float* out = (float*)d_out;

    size_t off = 0;
    auto alloc = [&](size_t bytes) -> void* {
        off = (off + 255) & ~(size_t)255;
        void* p = (char*)d_ws + off;
        off += bytes;
        return p;
    };
    // packed fragment sizes (f16 elements)
    const int PK2 = 5 * 2 * 512;    // K=64  -> 2 k-steps
    const int PK3 = 5 * 3 * 512;    // K=72  -> 3 k-steps
    const int PK27 = 5 * 27 * 512;  // K=864 -> 27 k-steps

    float*    deg  = (float*)alloc((size_t)NN * 4);
    float*    nrm  = (float*)alloc((size_t)NE * 4);
    float*    h    = (float*)alloc((size_t)NN * HID * 4);
    float*    h2   = (float*)alloc((size_t)NN * HID * 4);
    float*    agg  = (float*)alloc((size_t)NN * HID * 4);
    float*    hw   = (float*)alloc((size_t)NN * HID * 4);
    int*      bidx = (int*)alloc((size_t)NN * 4);
    float*    cnt  = (float*)alloc((size_t)NB * 4);
    float*    p1   = (float*)alloc((size_t)NB * 2 * HID * 4);
    unsigned* p2   = (unsigned*)alloc((size_t)NB * 2 * HID * 4);
    float*    xp   = (float*)alloc((size_t)NB * 6 * HID * 4);
    float*    stat = (float*)alloc(2 * HID * 4);
    _Float16* nwP  = (_Float16*)alloc((size_t)PK2 * 2);
    _Float16* ewP  = (_Float16*)alloc((size_t)PK2 * 2);
    _Float16* wiP  = (_Float16*)alloc((size_t)NLAY * PK3 * 2);
    _Float16* wrP  = (_Float16*)alloc((size_t)NLAY * PK3 * 2);
    _Float16* w1P  = (_Float16*)alloc((size_t)PK27 * 2);
    float* ef = hw;  // hw dead after last layer; reuse for emb features

    const int* rowp = ei;
    const int* colp = ei + NE;

    // degree -> d^-1/2 -> per-edge norm
    hipMemsetAsync(deg, 0, (size_t)NN * 4, stream);
    k_deg <<<(NE + 255) / 256, 256, 0, stream>>>(colp, deg);
    k_dis <<<(NN + 255) / 256, 256, 0, stream>>>(deg);
    k_norm<<<(NE + 255) / 256, 256, 0, stream>>>(rowp, colp, deg, nrm);

    // pack weights into WMMA B-fragment order (tiny, once per call)
    k_packB<<<(PK2 + 255) / 256, 256, 0, stream>>>(node_w, nwP, KNIN, HID, 2, PK2);
    k_packB<<<(PK2 + 255) / 256, 256, 0, stream>>>(emb_w,  ewP, KNIN, HID, 2, PK2);
    for (int l = 0; l < NLAY; ++l) {
        k_packB<<<(PK3 + 255) / 256, 256, 0, stream>>>(wi + (size_t)l * HID * HID,
                                                       wiP + (size_t)l * PK3, HID, HID, 3, PK3);
        k_packB<<<(PK3 + 255) / 256, 256, 0, stream>>>(wr + (size_t)l * HID * HID,
                                                       wrP + (size_t)l * PK3, HID, HID, 3, PK3);
    }
    k_packB<<<(PK27 + 255) / 256, 256, 0, stream>>>(w1, w1P, 12 * HID, HID, 27, PK27);

    const int gblocks = (NN + 15) / 16;
    // h0 = relu(x @ node_w + node_b)
    k_gemm16<<<gblocks, 160, 0, stream>>>(x, NN, KNIN, 2, nwP, node_b, nullptr, 1, h);

    for (int l = 0; l < NLAY; ++l) {
        // hw = h @ Wi[l]
        k_gemm16<<<gblocks, 160, 0, stream>>>(h, NN, HID, 3, wiP + (size_t)l * PK3,
                                              nullptr, nullptr, 0, hw);
        // agg = segment_sum(norm * hw[row], col)
        hipMemsetAsync(agg, 0, (size_t)NN * HID * 4, stream);
        k_edge_scatter<<<(NE + 7) / 8, 256, 0, stream>>>(rowp, colp, nrm, hw, agg);
        // h2 = relu(agg + h @ Wr[l] + bias[l])
        k_gemm16<<<gblocks, 160, 0, stream>>>(h, NN, HID, 3, wrP + (size_t)l * PK3,
                                              cb + l * HID, agg, 1, h2);
        // batchnorm (training stats) -> h
        hipMemsetAsync(stat, 0, 2 * HID * 4, stream);
        k_bn_stats<<<(NN + 63) / 64, 256, 0, stream>>>(h2, stat);
        k_bn_apply<<<(NN * HID + 255) / 256, 256, 0, stream>>>(h2, stat, gam + l * HID,
                                                               bet + l * HID, h);
    }

    // e = relu(emb @ emb_w + emb_b)
    k_gemm16<<<gblocks, 160, 0, stream>>>(emb, NN, KNIN, 2, ewP, emb_b, nullptr, 1, ef);

    // pooling path
    k_argmax<<<(NN + 7) / 8, 256, 0, stream>>>(assign, bidx);
    hipMemsetAsync(cnt, 0, (size_t)NB * 4, stream);
    hipMemsetAsync(p1,  0, (size_t)NB * 2 * HID * 4, stream);
    hipMemsetAsync(p2,  0, (size_t)NB * 2 * HID * 4, stream);
    k_pool<<<(NN + 7) / 8, 256, 0, stream>>>(h, ef, bidx, cnt, p1, p2);
    k_xp<<<(NB * 6 * HID + 255) / 256, 256, 0, stream>>>(cnt, p1, p2, xp);

    // final pair MLP
    k_pair_mlp<<<(NEP + 15) / 16, 160, 0, stream>>>(xp, pe, w1P, b1, w2, b2, out);
}
